// TTTMLP_30683246362768
// MI455X (gfx1250) — matrix operand order repaired and checked by reading, not compiled, not run on hardware
//
#include <hip/hip_runtime.h>
#include <math.h>

typedef __bf16 bf16;
typedef __attribute__((ext_vector_type(16))) __bf16 v16bf;
typedef __attribute__((ext_vector_type(8)))  float  v8f;
typedef int v4i_t __attribute__((vector_size(16)));
typedef __attribute__((address_space(3))) v4i_t v4i_lds_t;

#define DEV static __device__ __forceinline__
#define GELU_K 0.7978845608028654f

#if __has_builtin(__builtin_amdgcn_global_load_async_to_lds_b128) && \
    __has_builtin(__builtin_amdgcn_s_wait_asynccnt)
#define HAVE_ASYNC_LDS 1
#endif

DEV v8f wmma_bf(v16bf a, v16bf b, v8f c) {
  return __builtin_amdgcn_wmma_f32_16x16x32_bf16(false, a, false, b, (short)0, c, false, false);
}

DEV v8f v8f_splat(float x) { v8f r;
#pragma unroll
  for (int v = 0; v < 8; ++v) r[v] = x;
  return r; }

DEV float wsum(float v) {
#pragma unroll
  for (int o = 16; o > 0; o >>= 1) v += __shfl_xor(v, o, 32);
  return v;
}

DEV float gelu_f(float x) {
  float t = tanhf(GELU_K * (x + 0.044715f * x * x * x));
  return 0.5f * x * (1.0f + t);
}
DEV float gelu_bwd_f(float x) {
  float t = tanhf(GELU_K * (x + 0.044715f * x * x * x));
  return 0.5f * (1.0f + t) + 0.5f * x * (1.0f - t * t) * GELU_K * (1.0f + 3.0f * 0.044715f * x * x);
}

// K-offset map for 16-bit A fragments (16x32): element e of a lane
DEV int a_kmap(int e, int h8) { return 2 * h8 + e; }

// ---- fragment loaders (per CDNA5 WMMA VGPR layouts, wave32) ----
DEV v16bf ldA_row_f32(const float* S, int pitch, int m0, int k0) {
  int l = threadIdx.x & 31, r = l & 15, h8 = (l >> 4) * 8;
  const float* p = S + (m0 + r) * pitch + k0;
  v16bf a;
#pragma unroll
  for (int e = 0; e < 16; ++e) a[e] = (bf16)p[a_kmap(e, h8)];
  return a;
}
DEV v16bf ldA_row_bf16(const bf16* S, int pitch, int m0, int k0) {
  int l = threadIdx.x & 31, r = l & 15, h8 = (l >> 4) * 8;
  const bf16* p = S + (m0 + r) * pitch + k0;
  v16bf a;
#pragma unroll
  for (int e = 0; e < 16; ++e) a[e] = p[a_kmap(e, h8)];
  return a;
}
DEV v16bf ldB_k_bf16(const bf16* S, int pitch, int k0, int n0) { // B[k][n] = S[k][n]
  int l = threadIdx.x & 31, n = l & 15, kb = (l >> 4) * 16;
  v16bf b;
#pragma unroll
  for (int e = 0; e < 16; ++e) b[e] = S[(k0 + kb + e) * pitch + n0 + n];
  return b;
}
DEV v16bf ldB_k_bf16n(const bf16* S, int pitch, int k0, int n0) { // B[k][n] = -S[k][n]
  int l = threadIdx.x & 31, n = l & 15, kb = (l >> 4) * 16;
  v16bf b;
#pragma unroll
  for (int e = 0; e < 16; ++e) b[e] = (bf16)(-(float)S[(k0 + kb + e) * pitch + n0 + n]);
  return b;
}
DEV v16bf ldB_k_f32n(const float* S, int pitch, int k0, int n0) { // B[k][n] = -S[k][n]
  int l = threadIdx.x & 31, n = l & 15, kb = (l >> 4) * 16;
  v16bf b;
#pragma unroll
  for (int e = 0; e < 16; ++e) b[e] = (bf16)(-S[(k0 + kb + e) * pitch + n0 + n]);
  return b;
}
DEV v16bf ldB_n_bf16(const bf16* S, int pitch, int k0, int n0) { // B[k][n] = S[n][k]
  int l = threadIdx.x & 31, n = l & 15, kb = (l >> 4) * 16;
  const bf16* p = S + (n0 + n) * pitch + k0 + kb;
  v16bf b;
#pragma unroll
  for (int e = 0; e < 16; ++e) b[e] = p[e];
  return b;
}
DEV v16bf ldB_n_f32(const float* S, int pitch, int k0, int n0) {
  int l = threadIdx.x & 31, n = l & 15, kb = (l >> 4) * 16;
  const float* p = S + (n0 + n) * pitch + k0 + kb;
  v16bf b;
#pragma unroll
  for (int e = 0; e < 16; ++e) b[e] = (bf16)p[e];
  return b;
}
DEV void stC_f32(float* S, int pitch, int m0, int n0, v8f c) {
  int l = threadIdx.x & 31, n = l & 15, mh = (l >> 4) * 8;
#pragma unroll
  for (int v = 0; v < 8; ++v) S[(m0 + mh + v) * pitch + n0 + n] = c[v];
}
DEV void stC_bf16(bf16* S, int pitch, int m0, int n0, v8f c) {
  int l = threadIdx.x & 31, n = l & 15, mh = (l >> 4) * 8;
#pragma unroll
  for (int v = 0; v < 8; ++v) S[(m0 + mh + v) * pitch + n0 + n] = (bf16)c[v];
}
DEV v8f ldC_bf16(const bf16* S, int pitch, int m0, int n0) {
  int l = threadIdx.x & 31, n = l & 15, mh = (l >> 4) * 8;
  v8f c;
#pragma unroll
  for (int v = 0; v < 8; ++v) c[v] = (float)S[(m0 + mh + v) * pitch + n0 + n];
  return c;
}

// ======================================================================
// GEMM: C[m,n] = sum_k A[m,k] * W[n,k]   (M=8192, N=1024, K=1024)
// 128x64 block tile, K-chunk 64, ping-pong LDS, register-staged global
// loads overlapped with WMMA. blockIdx.z selects weight/dst (QKV in one
// launch). permute=1 writes output in [NC,B,H,CS,hd] scan layout.
// ======================================================================
__global__ __launch_bounds__(256)
void k_gemm_nt(const float* __restrict__ A,
               const float* __restrict__ Wa, const float* __restrict__ Wb,
               const float* __restrict__ Wc,
               float* __restrict__ Ca, float* __restrict__ Cb, float* __restrict__ Cc,
               int permute) {
  __shared__ bf16 As[2][128 * 72];
  __shared__ bf16 Bs[2][64 * 72];
  int z = blockIdx.z;
  const float* W = (z == 0) ? Wa : ((z == 1) ? Wb : Wc);
  float* C = (z == 0) ? Ca : ((z == 1) ? Cb : Cc);
  int tid = threadIdx.x, lane = tid & 31, wid = tid >> 5;
  int m0 = blockIdx.x * 128, n0 = blockIdx.y * 64;
  int wm = (wid & 3) * 32, wn = (wid >> 2) * 32;
  int ar = tid >> 1, ac = (tid & 1) * 32;       // A stage: 128 rows x 64 cols
  v8f acc[2][2];
#pragma unroll
  for (int i = 0; i < 2; ++i)
#pragma unroll
    for (int j = 0; j < 2; ++j) acc[i][j] = v8f_splat(0.f);

  float4 ra[8], rb[8];
  // prologue: stage chunk 0 into buffer 0
  {
    const float4* pa = (const float4*)(A + (size_t)(m0 + ar) * 1024 + ac);
#pragma unroll
    for (int j = 0; j < 8; ++j) ra[j] = pa[j];
    if (tid < 128) {
      const float4* pb = (const float4*)(W + (size_t)(n0 + (tid >> 1)) * 1024 + (tid & 1) * 32);
#pragma unroll
      for (int j = 0; j < 8; ++j) rb[j] = pb[j];
    }
    bf16* da = As[0] + ar * 72 + ac;
#pragma unroll
    for (int j = 0; j < 8; ++j) {
      da[j * 4 + 0] = (bf16)ra[j].x; da[j * 4 + 1] = (bf16)ra[j].y;
      da[j * 4 + 2] = (bf16)ra[j].z; da[j * 4 + 3] = (bf16)ra[j].w;
    }
    if (tid < 128) {
      bf16* db = Bs[0] + (tid >> 1) * 72 + (tid & 1) * 32;
#pragma unroll
      for (int j = 0; j < 8; ++j) {
        db[j * 4 + 0] = (bf16)rb[j].x; db[j * 4 + 1] = (bf16)rb[j].y;
        db[j * 4 + 2] = (bf16)rb[j].z; db[j * 4 + 3] = (bf16)rb[j].w;
      }
    }
  }
  __syncthreads();

  for (int kc = 0; kc < 16; ++kc) {
    int cur = kc & 1;
    bool more = (kc + 1) < 16;
    if (more) { // issue global loads for next chunk (wait lands after compute)
      int kk = (kc + 1) * 64;
      const float4* pa = (const float4*)(A + (size_t)(m0 + ar) * 1024 + kk + ac);
#pragma unroll
      for (int j = 0; j < 8; ++j) ra[j] = pa[j];
      if (tid < 128) {
        const float4* pb = (const float4*)(W + (size_t)(n0 + (tid >> 1)) * 1024 + kk + (tid & 1) * 32);
#pragma unroll
        for (int j = 0; j < 8; ++j) rb[j] = pb[j];
      }
    }
#pragma unroll
    for (int ks = 0; ks < 64; ks += 32) {
      v16bf a0 = ldA_row_bf16(As[cur], 72, wm, ks);
      v16bf a1 = ldA_row_bf16(As[cur], 72, wm + 16, ks);
      v16bf b0 = ldB_n_bf16(Bs[cur], 72, ks, wn);
      v16bf b1 = ldB_n_bf16(Bs[cur], 72, ks, wn + 16);
      acc[0][0] = wmma_bf(a0, b0, acc[0][0]);
      acc[0][1] = wmma_bf(a0, b1, acc[0][1]);
      acc[1][0] = wmma_bf(a1, b0, acc[1][0]);
      acc[1][1] = wmma_bf(a1, b1, acc[1][1]);
    }
    if (more) {
      int nxt = cur ^ 1;
      bf16* da = As[nxt] + ar * 72 + ac;
#pragma unroll
      for (int j = 0; j < 8; ++j) {
        da[j * 4 + 0] = (bf16)ra[j].x; da[j * 4 + 1] = (bf16)ra[j].y;
        da[j * 4 + 2] = (bf16)ra[j].z; da[j * 4 + 3] = (bf16)ra[j].w;
      }
      if (tid < 128) {
        bf16* db = Bs[nxt] + (tid >> 1) * 72 + (tid & 1) * 32;
#pragma unroll
        for (int j = 0; j < 8; ++j) {
          db[j * 4 + 0] = (bf16)rb[j].x; db[j * 4 + 1] = (bf16)rb[j].y;
          db[j * 4 + 2] = (bf16)rb[j].z; db[j * 4 + 3] = (bf16)rb[j].w;
        }
      }
    }
    __syncthreads();
  }

  int nn = lane & 15, mh = (lane >> 4) * 8;
#pragma unroll
  for (int i = 0; i < 2; ++i)
#pragma unroll
    for (int j = 0; j < 2; ++j) {
#pragma unroll
      for (int v = 0; v < 8; ++v) {
        int m = m0 + wm + i * 16 + mh + v;
        int n = n0 + wn + j * 16 + nn;
        float val = acc[i][j][v];
        if (!permute) {
          C[(size_t)m * 1024 + n] = val;
        } else {
          int b = m >> 11, l2 = m & 2047, nc = l2 >> 6, cs = l2 & 63;
          int h = n >> 6, cc = n & 63;
          C[(((nc * 4 + b) * 16 + h) * 64 + cs) * 64 + cc] = val;
        }
      }
    }
}

// ======================================================================
// l2norm + rope (in place) over [NC,B,H,CS,hd]; one row (hd=64) per wave.
// ======================================================================
__global__ __launch_bounds__(256)
void k_l2rope(float* __restrict__ XQr, float* __restrict__ XKr) {
  int lane = threadIdx.x & 31, wid = threadIdx.x >> 5;
  int row = blockIdx.x * 8 + wid;
  float* p = ((blockIdx.z == 0) ? XQr : XKr) + (size_t)row * 64;
  int cs = row & 63;
  float x0 = p[lane], x1 = p[lane + 32];
  float ss = wsum(x0 * x0 + x1 * x1);
  float inv = 1.f / fmaxf(sqrtf(ss), 1e-12f);
  x0 *= inv; x1 *= inv;
  float invf = powf(10000.f, -((float)lane) * (1.f / 32.f));
  float ang = (float)cs * invf;
  float cv = cosf(ang), sv = sinf(ang);
  p[lane] = x0 * cv - x1 * sv;
  p[lane + 32] = x1 * cv + x0 * sv;
}

// XVt = g*(dv-mu)/(std_ddof1+1e-8) + b + XK  (in place over XV)
__global__ __launch_bounds__(256)
void k_xvt(float* __restrict__ XVr, const float* __restrict__ XKr,
           const float* __restrict__ tnw, const float* __restrict__ tnb) {
  int lane = threadIdx.x & 31, wid = threadIdx.x >> 5;
  int row = blockIdx.x * 8 + wid;
  int h = (row >> 6) & 15;
  float* pv = XVr + (size_t)row * 64;
  const float* pk = XKr + (size_t)row * 64;
  float k0 = pk[lane], k1 = pk[lane + 32];
  float d0 = pv[lane] - k0, d1 = pv[lane + 32] - k1;
  float mu = wsum(d0 + d1) * (1.f / 64.f);
  float c0 = d0 - mu, c1 = d1 - mu;
  float sd = sqrtf(wsum(c0 * c0 + c1 * c1) * (1.f / 63.f));
  float inv = 1.f / (sd + 1e-8f);
  pv[lane] = tnw[h * 64 + lane] * c0 * inv + tnb[h * 64 + lane] + k0;
  pv[lane + 32] = tnw[h * 64 + lane + 32] * c1 * inv + tnb[h * 64 + lane + 32] + k1;
}

// eta[nc,b,h,cs] = sigmoid(x[b,l]·lrw[h] + lrb[h]) / 64
__global__ __launch_bounds__(256)
void k_eta(const float* __restrict__ x, const float* __restrict__ lrw,
           const float* __restrict__ lrb, float* __restrict__ ETA) {
  __shared__ float xs[1024];
  int tid = threadIdx.x, lane = tid & 31, wid = tid >> 5;
  int row = blockIdx.x;
  ((float4*)xs)[tid] = ((const float4*)(x + (size_t)row * 1024))[tid];
  __syncthreads();
  int b = row >> 11, l2 = row & 2047, nc = l2 >> 6, cs = l2 & 63;
#pragma unroll
  for (int hh = 0; hh < 2; ++hh) {
    int h = wid * 2 + hh;
    const float* w = lrw + h * 1024;
    float s = 0.f;
    for (int k = lane; k < 1024; k += 32) s += xs[k] * w[k];
    s = wsum(s);
    if (lane == 0) {
      float v = 1.f / (1.f + expf(-(s + lrb[h])));
      ETA[((nc * 4 + b) * 16 + h) * 64 + cs] = v * (1.f / 64.f);
    }
  }
}

// ======================================================================
// TTT scan: one block per (b,h). State stored transposed so that every
// hot B-fragment is a contiguous 16B LDS read:
//   W1t[f=256][d=64]  (pitch 72)   W2t[d=64][f=256]  (pitch 264)
//   G1t[f=256][j=64]  (pitch 72)   G2r[j][d] + G2t[d][j] (pitch 72)
// 274,432 B dynamic LDS (< 320KB/WGP).
// ======================================================================
__global__ __launch_bounds__(256)
void k_ttt_scan(const float* __restrict__ XQr, const float* __restrict__ XKr,
                const float* __restrict__ XVr, const float* __restrict__ ETA,
                const float* __restrict__ W1g, const float* __restrict__ b1g,
                const float* __restrict__ W2g, const float* __restrict__ b2g,
                const float* __restrict__ tnw, const float* __restrict__ tnb,
                float* __restrict__ Y) {
  extern __shared__ char smem[];
  float* xq = (float*)smem;          // 64 x 68 f32
  float* xk = xq + 64 * 68;
  float* xv = xk + 64 * 68;
  float* zb = xv + 64 * 68;          // 64 x 68 (Z2 / Z2_bar f32)
  float* b1s = zb + 64 * 68;         // 256
  float* b2s = b1s + 256;            // 64
  float* etas = b2s + 64;            // 64
  float* glnv = etas + 64;           // 64
  float* blnv = glnv + 64;           // 64
  bf16* W1t = (bf16*)(blnv + 64);    // 256 x 72  W1t[f][d]
  bf16* W2t = W1t + 256 * 72;        // 64 x 264  W2t[d][f]
  bf16* Z1b = W2t + 64 * 264;        // 64 x 264  (Z1 -> X2_bar)
  bf16* X2b = Z1b + 64 * 264;        // 64 x 264
  bf16* G1t = X2b + 64 * 264;        // 256 x 72  G1t[f][j] (eta-scaled gZ1)
  bf16* A1b = G1t + 256 * 72;        // 64 x 72   (-tril(Attn+1))
  bf16* G2r = A1b + 64 * 72;         // 64 x 72   gZ2[j][d]
  bf16* G2t = G2r + 64 * 72;         // 64 x 72   gZ2^T[d][j]

  const int tid = threadIdx.x, lane = tid & 31, wid = tid >> 5;
  const int bb = blockIdx.x >> 4, hh = blockIdx.x & 15;

  for (int i = tid; i < 64 * 256; i += 256) { // W1g[h][d][f] -> W1t[f][d]
    int d = i >> 8, f = i & 255;
    W1t[f * 72 + d] = (bf16)W1g[hh * 16384 + i];
  }
  for (int i = tid; i < 256 * 64; i += 256) { // W2g[h][f][d] -> W2t[d][f]
    int f = i >> 6, d = i & 63;
    W2t[d * 264 + f] = (bf16)W2g[hh * 16384 + i];
  }
  b1s[tid] = b1g[hh * 256 + tid];
  if (tid < 64) {
    b2s[tid] = b2g[hh * 64 + tid];
    glnv[tid] = tnw[hh * 64 + tid];
    blnv[tid] = tnb[hh * 64 + tid];
  }
  __syncthreads();

  for (int nc = 0; nc < 32; ++nc) {
    const int base = ((nc * 4 + bb) * 16 + hh) << 12;
    { // phase 0: load chunk (async global->LDS when available)
      int off = tid * 16, r = off >> 6, c = off & 63;
      int lo = (r * 68 + c) * 4;
#ifdef HAVE_ASYNC_LDS
#pragma unroll
      for (int j = 0; j < 4; ++j) {
        __builtin_amdgcn_global_load_async_to_lds_b128(
            (v4i_t*)(XQr + base + off + j * 4),
            (v4i_lds_t*)((char*)xq + lo + j * 16), 0, 0);
        __builtin_amdgcn_global_load_async_to_lds_b128(
            (v4i_t*)(XKr + base + off + j * 4),
            (v4i_lds_t*)((char*)xk + lo + j * 16), 0, 0);
        __builtin_amdgcn_global_load_async_to_lds_b128(
            (v4i_t*)(XVr + base + off + j * 4),
            (v4i_lds_t*)((char*)xv + lo + j * 16), 0, 0);
      }
      __builtin_amdgcn_s_wait_asynccnt(0);
#else
      const float4* q4 = (const float4*)(XQr + base + off);
      const float4* k4 = (const float4*)(XKr + base + off);
      const float4* v4 = (const float4*)(XVr + base + off);
#pragma unroll
      for (int j = 0; j < 4; ++j) {
        ((float4*)((char*)xq + lo))[j] = q4[j];
        ((float4*)((char*)xk + lo))[j] = k4[j];
        ((float4*)((char*)xv + lo))[j] = v4[j];
      }
#endif
      if (tid < 64) etas[tid] = ETA[((nc * 4 + bb) * 16 + hh) * 64 + tid];
    }
    __syncthreads();

    // phase 1: Z1 = xk @ W1 + b1 (64x256);  B[d][f] = W1t[f][d] (contig)
    for (int t = wid; t < 64; t += 8) {
      int tm = (t >> 4) * 16, tn = (t & 15) * 16;
      v8f acc = v8f_splat(b1s[tn + (lane & 15)]);
#pragma unroll
      for (int k0 = 0; k0 < 64; k0 += 32)
        acc = wmma_bf(ldA_row_f32(xk, 68, tm, k0), ldB_n_bf16(W1t, 72, k0, tn), acc);
      stC_bf16(Z1b, 264, tm, tn, acc);
    }
    __syncthreads();

    // phase 2: X2 = gelu(Z1)
    for (int i = tid; i < 64 * 256; i += 256) {
      int idx = (i >> 8) * 264 + (i & 255);
      X2b[idx] = (bf16)gelu_f((float)Z1b[idx]);
    }
    __syncthreads();

    // phase 3: Z2 = X2 @ W2 + b2 -> zb;  B[f][d] = W2t[d][f] (contig)
    for (int t = wid; t < 16; t += 8) {
      int tm = (t >> 2) * 16, tn = (t & 3) * 16;
      v8f acc = v8f_splat(b2s[tn + (lane & 15)]);
#pragma unroll
      for (int k0 = 0; k0 < 256; k0 += 32)
        acc = wmma_bf(ldA_row_bf16(X2b, 264, tm, k0), ldB_n_bf16(W2t, 264, k0, tn), acc);
      stC_f32(zb, 68, tm, tn, acc);
    }
    __syncthreads();

    // phase 4: gZ2 = ln_l2_bwd(Z2, xv-xk) (unscaled) -> G2r and G2t
    for (int rr = 0; rr < 8; ++rr) {
      int r = wid * 8 + rr;
      float z0 = zb[r * 68 + lane], z1 = zb[r * 68 + lane + 32];
      float t0 = xv[r * 68 + lane] - xk[r * 68 + lane];
      float t1 = xv[r * 68 + lane + 32] - xk[r * 68 + lane + 32];
      float mu = wsum(z0 + z1) * (1.f / 64.f);
      float c0 = z0 - mu, c1 = z1 - mu;
      float var = wsum(c0 * c0 + c1 * c1) * (1.f / 64.f);
      float rstd = rsqrtf(var + 1e-6f);
      float xh0 = c0 * rstd, xh1 = c1 * rstd;
      float g0 = glnv[lane], g1 = glnv[lane + 32];
      float gy0 = (g0 * xh0 + blnv[lane] - t0) * g0;
      float gy1 = (g1 * xh1 + blnv[lane + 32] - t1) * g1;
      float sg = wsum(gy0 + gy1);
      float sx = wsum(gy0 * xh0 + gy1 * xh1);
      float q0 = (64.f * gy0 - sg - xh0 * sx) * (rstd * (1.f / 64.f));
      float q1 = (64.f * gy1 - sg - xh1 * sx) * (rstd * (1.f / 64.f));
      G2r[r * 72 + lane] = (bf16)q0;
      G2r[r * 72 + lane + 32] = (bf16)q1;
      G2t[lane * 72 + r] = (bf16)q0;
      G2t[(lane + 32) * 72 + r] = (bf16)q1;
    }
    __syncthreads();

    // phase 5: G1 = (gZ2 @ W2^T) * gelu_bwd(Z1) * eta_row -> G1t (transposed)
    for (int t = wid; t < 64; t += 8) {
      int tm = (t >> 4) * 16, tn = (t & 15) * 16;
      v8f acc = v8f_splat(0.f);
#pragma unroll
      for (int k0 = 0; k0 < 64; k0 += 32)
        acc = wmma_bf(ldA_row_bf16(G2r, 72, tm, k0), ldB_k_bf16(W2t, 264, k0, tn), acc);
      int nn = lane & 15, mh = (lane >> 4) * 8;
#pragma unroll
      for (int v = 0; v < 8; ++v) {
        int m = tm + mh + v, n = tn + nn; // m=j, n=f
        float g = acc[v] * gelu_bwd_f((float)Z1b[m * 264 + n]) * etas[m];
        G1t[n * 72 + m] = (bf16)g;
      }
    }
    __syncthreads();

    // phase 5b: scale gZ2 copies by eta
    for (int i = tid; i < 64 * 64; i += 256) {
      int r = i >> 6, c = i & 63;
      G2r[r * 72 + c] = (bf16)((float)G2r[r * 72 + c] * etas[r]);
      G2t[r * 72 + c] = (bf16)((float)G2t[r * 72 + c] * etas[c]);
    }
    __syncthreads();

    // phase 6: A1 = -(tril(xq@xk^T) + tril_ones)
    for (int t = wid; t < 16; t += 8) {
      int tm = (t >> 2) * 16, tn = (t & 3) * 16;
      v8f acc = v8f_splat(0.f);
#pragma unroll
      for (int k0 = 0; k0 < 64; k0 += 32)
        acc = wmma_bf(ldA_row_f32(xq, 68, tm, k0), ldB_n_f32(xk, 68, k0, tn), acc);
      int nn = lane & 15, mh = (lane >> 4) * 8;
#pragma unroll
      for (int v = 0; v < 8; ++v) {
        int m = tm + mh + v, n = tn + nn;
        A1b[m * 72 + n] = (bf16)((n <= m) ? -(acc[v] + 1.f) : 0.f);
      }
    }
    __syncthreads();

    // phase 7: X2_bar = gelu(xq@W1 + b1 + A1@G1) -> Z1b
    for (int t = wid; t < 64; t += 8) {
      int tm = (t >> 4) * 16, tn = (t & 15) * 16;
      v8f acc = v8f_splat(b1s[tn + (lane & 15)]);
#pragma unroll
      for (int k0 = 0; k0 < 64; k0 += 32) {
        acc = wmma_bf(ldA_row_f32(xq, 68, tm, k0), ldB_n_bf16(W1t, 72, k0, tn), acc);
        acc = wmma_bf(ldA_row_bf16(A1b, 72, tm, k0), ldB_n_bf16(G1t, 72, k0, tn), acc);
      }
      int nn = lane & 15, mh = (lane >> 4) * 8;
#pragma unroll
      for (int v = 0; v < 8; ++v)
        Z1b[(tm + mh + v) * 264 + tn + nn] = (bf16)gelu_f(acc[v]);
    }
    __syncthreads();

    // phase 9: A1 = -(tril(X2_bar@X2^T) + tril_ones)
    for (int t = wid; t < 16; t += 8) {
      int tm = (t >> 2) * 16, tn = (t & 3) * 16;
      v8f acc = v8f_splat(0.f);
#pragma unroll
      for (int k0 = 0; k0 < 256; k0 += 32)
        acc = wmma_bf(ldA_row_bf16(Z1b, 264, tm, k0), ldB_n_bf16(X2b, 264, k0, tn), acc);
      int nn = lane & 15, mh = (lane >> 4) * 8;
#pragma unroll
      for (int v = 0; v < 8; ++v) {
        int m = tm + mh + v, n = tn + nn;
        A1b[m * 72 + n] = (bf16)((n <= m) ? -(acc[v] + 1.f) : 0.f);
      }
    }
    __syncthreads();

    // phase 10: Z2_bar = X2_bar@W2 + b2 + A1@G2 -> zb
    for (int t = wid; t < 16; t += 8) {
      int tm = (t >> 2) * 16, tn = (t & 3) * 16;
      v8f acc = v8f_splat(b2s[tn + (lane & 15)]);
#pragma unroll
      for (int k0 = 0; k0 < 256; k0 += 32)
        acc = wmma_bf(ldA_row_bf16(Z1b, 264, tm, k0), ldB_n_bf16(W2t, 264, k0, tn), acc);
#pragma unroll
      for (int k0 = 0; k0 < 64; k0 += 32)
        acc = wmma_bf(ldA_row_bf16(A1b, 72, tm, k0), ldB_n_bf16(G2t, 72, k0, tn), acc);
      stC_f32(zb, 68, tm, tn, acc);
    }
    __syncthreads();

    // phase 10b: out = xq + ln_fwd(Z2_bar) -> Y [B,L,D]
    for (int rr = 0; rr < 8; ++rr) {
      int r = wid * 8 + rr;
      float z0 = zb[r * 68 + lane], z1 = zb[r * 68 + lane + 32];
      float mu = wsum(z0 + z1) * (1.f / 64.f);
      float c0 = z0 - mu, c1 = z1 - mu;
      float var = wsum(c0 * c0 + c1 * c1) * (1.f / 64.f);
      float rstd = rsqrtf(var + 1e-6f);
      int yi = ((bb * 2048) + nc * 64 + r) * 1024 + hh * 64;
      Y[yi + lane] = glnv[lane] * c0 * rstd + blnv[lane] + xq[r * 68 + lane];
      Y[yi + lane + 32] = glnv[lane + 32] * c1 * rstd + blnv[lane + 32] + xq[r * 68 + lane + 32];
    }
    __syncthreads();

    // phase 11: W1t[f][d] -= sum_j G1t[f][j] * xk[j][d] ; b1 -= colsum(G1)
    for (int t = wid; t < 64; t += 8) {
      int tm = (t >> 2) * 16, tn = (t & 3) * 16; // tm over f(256), tn over d(64)
      v8f acc = ldC_bf16(W1t, 72, tm, tn);
#pragma unroll
      for (int k0 = 0; k0 < 64; k0 += 32)
        acc = wmma_bf(ldA_row_bf16(G1t, 72, tm, k0), ldB_k_f32n(xk, 68, k0, tn), acc);
      stC_bf16(W1t, 72, tm, tn, acc);
    }
    {
      float s = 0.f;
      for (int j = 0; j < 64; ++j) s += (float)G1t[tid * 72 + j];
      b1s[tid] -= s;
    }
    __syncthreads();

    // phase 12: W2t[d][f] -= sum_j G2t[d][j] * X2[j][f] ; b2 -= colsum(G2)
    for (int t = wid; t < 64; t += 8) {
      int tm = (t >> 4) * 16, tn = (t & 15) * 16; // tm over d(64), tn over f(256)
      v8f acc = ldC_bf16(W2t, 264, tm, tn);
#pragma unroll
      for (int k0 = 0; k0 < 64; k0 += 32)
        acc = wmma_bf(ldA_row_bf16(G2t, 72, tm, k0), ldB_k_bf16n(X2b, 264, k0, tn), acc);
      stC_bf16(W2t, 264, tm, tn, acc);
    }
    if (tid < 64) {
      float s = 0.f;
      for (int j = 0; j < 64; ++j) s += (float)G2t[tid * 72 + j];
      b2s[tid] -= s;
    }
    __syncthreads();
  }
}

// post-norm over D=1024 per token
__global__ __launch_bounds__(256)
void k_postnorm(const float* __restrict__ Y, const float* __restrict__ pw,
                const float* __restrict__ pb, float* __restrict__ YN) {
  __shared__ float red[16];
  int tid = threadIdx.x;
  int row = blockIdx.x;
  const float* yr = Y + (size_t)row * 1024;
  float s = 0.f, q = 0.f;
  for (int k = tid; k < 1024; k += 256) { float v = yr[k]; s += v; q += v * v; }
  s = wsum(s); q = wsum(q);
  if ((tid & 31) == 0) { red[tid >> 5] = s; red[8 + (tid >> 5)] = q; }
  __syncthreads();
  if (tid == 0) {
    float S = 0.f, Q = 0.f;
    for (int i = 0; i < 8; ++i) { S += red[i]; Q += red[8 + i]; }
    float mu = S * (1.f / 1024.f);
    float var = Q * (1.f / 1024.f) - mu * mu;
    red[0] = mu; red[1] = rsqrtf(var + 1e-6f);
  }
  __syncthreads();
  float mu = red[0], rstd = red[1];
  float* yo = YN + (size_t)row * 1024;
  for (int k = tid; k < 1024; k += 256) yo[k] = pw[k] * (yr[k] - mu) * rstd + pb[k];
}

extern "C" void kernel_launch(void* const* d_in, const int* in_sizes, int n_in,
                              void* d_out, int out_size, void* d_ws, size_t ws_size,
                              hipStream_t stream) {
  const float* x = (const float*)d_in[0];
  const float* Wq = (const float*)d_in[1];
  const float* Wk = (const float*)d_in[2];
  const float* Wv = (const float*)d_in[3];
  const float* Wo = (const float*)d_in[4];
  const float* W1g = (const float*)d_in[5];
  const float* b1g = (const float*)d_in[6];
  const float* W2g = (const float*)d_in[7];
  const float* b2g = (const float*)d_in[8];
  const float* tnw = (const float*)d_in[9];
  const float* tnb = (const float*)d_in[10];
  const float* lrw = (const float*)d_in[11];
  const float* lrb = (const float*)d_in[12];
  const float* pnw = (const float*)d_in[13];
  const float* pnb = (const float*)d_in[14];
  float* out = (float*)d_out;

  float* ws = (float*)d_ws;
  const size_t NE = (size_t)8192 * 1024;
  float* XQr = ws;
  float* XKr = XQr + NE;
  float* XVr = XKr + NE;
  float* ETA = XVr + NE;
  float* Y = ETA + 131072;
  float* YN = Y + NE;

  // QKV projections (store in [NC,B,H,CS,hd] scan layout)
  k_gemm_nt<<<dim3(64, 16, 3), 256, 0, stream>>>(x, Wq, Wk, Wv, XQr, XKr, XVr, 1);
  // l2norm + rope on Q and K
  k_l2rope<<<dim3(16384, 1, 2), 256, 0, stream>>>(XQr, XKr);
  // XV whitening target (uses roped XK)
  k_xvt<<<dim3(16384), 256, 0, stream>>>(XVr, XKr, tnw, tnb);
  // learning rates
  k_eta<<<dim3(8192), 256, 0, stream>>>(x, lrw, lrb, ETA);
  // sequential TTT scan, one block per (b,h), 274,432 B LDS
  (void)hipFuncSetAttribute(reinterpret_cast<const void*>(k_ttt_scan),
                            hipFuncAttributeMaxDynamicSharedMemorySize, 274432);
  k_ttt_scan<<<dim3(64), 256, 274432, stream>>>(XQr, XKr, XVr, ETA, W1g, b1g, W2g, b2g,
                                                tnw, tnb, Y);
  // post-norm + output projection
  k_postnorm<<<dim3(8192), 256, 0, stream>>>(Y, pnw, pnb, YN);
  k_gemm_nt<<<dim3(64, 16, 1), 256, 0, stream>>>(YN, Wo, Wo, Wo, out, out, out, 0);
}
